// SAGPool_36816459661883
// MI455X (gfx1250) — compile-verified, matrix operand order checked by reading, not run on hardware
//
#include <hip/hip_runtime.h>
#include <hip/hip_bf16.h>

// Problem constants (match reference)
#define NNODES 32768   // G*N
#define GG     128
#define NPG    256
#define FIN    128
#define HH     256
#define EE     524288
#define CC     10

typedef __bf16   bf16x16 __attribute__((ext_vector_type(16)));
typedef float    f32x8   __attribute__((ext_vector_type(8)));
typedef unsigned u32x8   __attribute__((ext_vector_type(8)));

// pack two f32 into one dword of bf16 pairs (lowers to v_cvt_pk_bf16_f32)
__device__ inline unsigned pk2(float x, float y) {
  __bf16 a = (__bf16)x, b = (__bf16)y;
  unsigned short ua = __builtin_bit_cast(unsigned short, a);
  unsigned short ub = __builtin_bit_cast(unsigned short, b);
  return (unsigned)ua | ((unsigned)ub << 16);
}

// ---------------------------------------------------------------------------
// GEMM: C[M,N] = A[M,K] @ W[K,N] (+bias)(+relu), bf16 WMMA, f32 accumulate.
// Block tile 128x64x32, 8 waves; wave w owns rows [w*16, w*16+16) and all 64
// columns -> one A fragment reused across 4 v_wmma_f32_16x16x32_bf16 per
// k-step. A/B tiles staged through LDS as packed-bf16 dwords; B stored
// transposed [n][k] so each fragment K-pair is one aligned ds_load_b32.
// Fragment layouts per CDNA5 ISA: lane%16 = row(A)/col(B); VGPR v, half:
//   kpair = (v/4)*16 + (lane/16)*8 + (v%4)*2
// C 16x16 f32: VGPR r -> row r (lanes 0-15) / r+8 (lanes 16-31).
// Requires M%128==0, N%64==0, K%32==0 (true for all call sites).
// ---------------------------------------------------------------------------
#define BM 128
#define BN 64
#define BK 32

__global__ __launch_bounds__(256) void gemm_wmma(
    const float* __restrict__ A, const float* __restrict__ W,
    const float* __restrict__ bias, float* __restrict__ C,
    int M, int N, int K, int relu) {
  __shared__ unsigned Als[BM * (BK / 2)];   // 128x32 bf16, K-pairs packed
  __shared__ unsigned Bls[BN * (BK / 2)];   // 64x32 bf16, [n][k] K-pairs packed

  const int tid  = threadIdx.x;
  const int lane = tid & 31;
  const int wv   = tid >> 5;                // wave id 0..7 -> 16-row strip
  const int half = lane >> 4;
  const int l16  = lane & 15;
  const int m0   = blockIdx.x * BM;
  const int n0   = blockIdx.y * BN;

  f32x8 acc[4] = {};

  for (int k0 = 0; k0 < K; k0 += BK) {
    __syncthreads();                        // protect LDS from previous k-step
    // ---- stage A tile: 128x32 f32 -> bf16 (4 coalesced float4 per thread)
#pragma unroll
    for (int c = 0; c < 4; ++c) {
      int idx  = c * 256 + tid;             // 1024 float4 chunks
      int row  = idx >> 3;                  // 8 float4 per row
      int col4 = (idx & 7) << 2;
      const float* ap = A + (size_t)(m0 + row) * K + k0 + col4;
      float4 v = *(const float4*)ap;
      Als[row * 16 + (col4 >> 1)]     = pk2(v.x, v.y);
      Als[row * 16 + (col4 >> 1) + 1] = pk2(v.z, v.w);
      if (k0 + BK < K) __builtin_prefetch(ap + BK, 0, 1);  // next k-panel
    }
    // ---- stage B tile: 32x64 f32 -> bf16 transposed to [n][k]
    {
      __bf16* Bb = (__bf16*)Bls;
#pragma unroll
      for (int c = 0; c < 2; ++c) {
        int idx  = c * 256 + tid;           // 512 float4 chunks
        int kr   = idx >> 4;                // 16 float4 per k-row
        int col4 = (idx & 15) << 2;
        float4 v = *(const float4*)(W + (size_t)(k0 + kr) * N + n0 + col4);
        Bb[(col4 + 0) * BK + kr] = (__bf16)v.x;
        Bb[(col4 + 1) * BK + kr] = (__bf16)v.y;
        Bb[(col4 + 2) * BK + kr] = (__bf16)v.z;
        Bb[(col4 + 3) * BK + kr] = (__bf16)v.w;
      }
    }
    __syncthreads();

    // ---- A fragment (8 dwords) for this wave's 16-row strip
    u32x8 av;
#pragma unroll
    for (int v = 0; v < 8; ++v) {
      int kp = ((v >> 2) << 4) + (half << 3) + ((v & 3) << 1);
      av[v] = Als[(wv * 16 + l16) * 16 + (kp >> 1)];
    }
    bf16x16 afrag = __builtin_bit_cast(bf16x16, av);

    // ---- 4 WMMAs reuse afrag across the 64-column strip
#pragma unroll
    for (int q = 0; q < 4; ++q) {
      u32x8 bv;
#pragma unroll
      for (int v = 0; v < 8; ++v) {
        int kp = ((v >> 2) << 4) + (half << 3) + ((v & 3) << 1);
        bv[v] = Bls[(q * 16 + l16) * 16 + (kp >> 1)];
      }
      bf16x16 bfrag = __builtin_bit_cast(bf16x16, bv);
      acc[q] = __builtin_amdgcn_wmma_f32_16x16x32_bf16(
          false, afrag, false, bfrag, (short)0, acc[q], false, false);
    }
  }

  // ---- epilogue: bias + optional relu, f32 out
#pragma unroll
  for (int q = 0; q < 4; ++q) {
    int col = n0 + q * 16 + l16;
    float bc = bias ? bias[col] : 0.f;
#pragma unroll
    for (int r = 0; r < 8; ++r) {
      int row = m0 + wv * 16 + r + half * 8;
      float v = acc[q][r] + bc;
      if (relu) v = v > 0.f ? v : 0.f;
      C[(size_t)row * N + col] = v;
    }
  }
}

// ---------------------------------------------------------------------------
// Elementwise / graph helpers
// ---------------------------------------------------------------------------
__global__ void fill_f32(float* p, float v, int n) {
  int i = blockIdx.x * blockDim.x + threadIdx.x;
  if (i < n) p[i] = v;
}

__global__ void copy_f32(float* d, const float* s, int n) {
  int i = blockIdx.x * blockDim.x + threadIdx.x;
  if (i < n) d[i] = s[i];
}

// deg[dst] += emask (deg pre-initialized to nmask for the self-loop term)
__global__ void deg_edges(const int* __restrict__ dst, const float* __restrict__ em,
                          float* __restrict__ deg, int e) {
  int i = blockIdx.x * blockDim.x + threadIdx.x;
  if (i >= e) return;
  float w = em[i];
  if (w != 0.f) atomicAdd(deg + dst[i], w);
}

__global__ void dis_kernel(const float* __restrict__ deg, float* __restrict__ dis, int n) {
  int i = blockIdx.x * blockDim.x + threadIdx.x;
  if (i >= n) return;
  float d = deg[i];
  dis[i] = d > 0.f ? rsqrtf(fmaxf(d, 1e-12f)) : 0.f;
}

__global__ void norm_kernel(const int* __restrict__ src, const int* __restrict__ dst,
                            const float* __restrict__ em, const float* __restrict__ dis,
                            float* __restrict__ nrm, int e) {
  int i = blockIdx.x * blockDim.x + threadIdx.x;
  if (i >= e) return;
  nrm[i] = dis[src[i]] * dis[dst[i]] * em[i];
}

// out[dst] += h[src] * coef   (one thread per edge x float4 chunk, H = 256)
__global__ void aggregate(const float* __restrict__ h, const int* __restrict__ src,
                          const int* __restrict__ dst, const float* __restrict__ coef,
                          float* __restrict__ out, int e) {
  int idx = blockIdx.x * blockDim.x + threadIdx.x;   // e * (HH/4) threads
  int eI = idx >> 6;
  if (eI >= e) return;
  float w = coef[eI];
  if (w == 0.f) return;
  int f4 = (idx & 63) << 2;
  float4 v = *(const float4*)(h + (size_t)src[eI] * HH + f4);
  float* o = out + (size_t)dst[eI] * HH + f4;
  atomicAdd(o + 0, v.x * w);
  atomicAdd(o + 1, v.y * w);
  atomicAdd(o + 2, v.z * w);
  atomicAdd(o + 3, v.w * w);
}

// out = relu(agg + hw * dis^2 * nmask + b)   (agg already in out)
__global__ void gcn_epi(const float* __restrict__ hw, const float* __restrict__ dis,
                        const float* __restrict__ nmask, const float* __restrict__ b,
                        float* __restrict__ out) {
  int idx = blockIdx.x * blockDim.x + threadIdx.x;   // NNODES*HH threads
  int n = idx >> 8;
  int f = idx & 255;
  float d = dis[n];
  float v = out[idx] + hw[idx] * d * d * nmask[n] + b[f];
  out[idx] = v > 0.f ? v : 0.f;
}

// score[n] = nmask>0 ? sum_f(agg*w_rel + x*w_root) + b_rel : -inf  (one wave/node)
__global__ void score_kernel(const float* __restrict__ agg, const float* __restrict__ x,
                             const float* __restrict__ wrel, const float* __restrict__ brel,
                             const float* __restrict__ wroot, const float* __restrict__ nmask,
                             float* __restrict__ score) {
  int lane = threadIdx.x & 31;
  int node = blockIdx.x * 8 + (threadIdx.x >> 5);
  if (node >= NNODES) return;
  float s = 0.f;
  for (int f = lane; f < HH; f += 32)
    s += agg[(size_t)node * HH + f] * wrel[f] + x[(size_t)node * HH + f] * wroot[f];
  for (int o = 16; o > 0; o >>= 1) s += __shfl_down(s, o, 32);
  if (lane == 0)
    score[node] = nmask[node] > 0.f ? s + brel[0] : -__builtin_inff();
}

// per-graph top-k via 256-element bitonic sort (descending, lower-idx tiebreak)
__global__ void topk_kernel(const float* __restrict__ score, float* __restrict__ newmask, int k) {
  __shared__ float sc[NPG];
  __shared__ int   si[NPG];
  int t = threadIdx.x;
  int g = blockIdx.x;
  sc[t] = score[g * NPG + t];
  si[t] = t;
  __syncthreads();
  for (int ksz = 2; ksz <= NPG; ksz <<= 1) {
    for (int j = ksz >> 1; j > 0; j >>= 1) {
      int ixj = t ^ j;
      if (ixj > t) {
        float a = sc[t], b = sc[ixj];
        int ia = si[t], ib = si[ixj];
        bool dirUp = ((t & ksz) == 0);
        bool bBetter = (b > a) || (b == a && ib < ia);
        if (dirUp ? bBetter : !bBetter) {
          sc[t] = b; sc[ixj] = a; si[t] = ib; si[ixj] = ia;
        }
      }
      __syncthreads();
    }
  }
  newmask[g * NPG + si[t]] = (t < k) ? 1.f : 0.f;
}

__global__ void apply_pool(float* __restrict__ x, const float* __restrict__ score,
                           const float* __restrict__ nm) {
  int idx = blockIdx.x * blockDim.x + threadIdx.x;   // NNODES*HH
  int n = idx >> 8;
  x[idx] = x[idx] * tanhf(score[n]) * nm[n];
}

__global__ void update_emask(const int* __restrict__ src, const int* __restrict__ dst,
                             const float* __restrict__ nm, float* __restrict__ em, int e) {
  int i = blockIdx.x * blockDim.x + threadIdx.x;
  if (i >= e) return;
  em[i] = (em[i] != 0.f && nm[src[i]] > 0.f && nm[dst[i]] > 0.f) ? 1.f : 0.f;
}

// z[g, 0:768] (+)= [mean | max | sum] over masked nodes; one block/graph, thread=feature
__global__ void readout_kernel(const float* __restrict__ x, const float* __restrict__ nm,
                               float* __restrict__ z, int add) {
  int g = blockIdx.x, f = threadIdx.x;
  float s = 0.f, mx = -__builtin_inff(), cnt = 0.f;
  for (int n = 0; n < NPG; ++n) {
    float m = nm[g * NPG + n];
    if (m > 0.f) {
      float v = x[(size_t)(g * NPG + n) * HH + f];
      s += v;
      mx = fmaxf(mx, v);
      cnt += 1.f;
    }
  }
  float mean = s / cnt;
  float* zr = z + (size_t)g * (3 * HH);
  if (add) {
    zr[f] += mean; zr[HH + f] += mx; zr[2 * HH + f] += s;
  } else {
    zr[f] = mean;  zr[HH + f] = mx;  zr[2 * HH + f] = s;
  }
}

// logits[g,c] = zt2[g,:] @ fc3_W[:,c] + fc3_b[c]   (K=128, N=10)
__global__ void fc3_kernel(const float* __restrict__ zt2, const float* __restrict__ W,
                           const float* __restrict__ b, float* __restrict__ logits) {
  int i = blockIdx.x * blockDim.x + threadIdx.x;
  if (i >= GG * CC) return;
  int g = i / CC, c = i % CC;
  float s = b[c];
  for (int j = 0; j < HH / 2; ++j) s += zt2[g * (HH / 2) + j] * W[j * CC + c];
  logits[i] = s;
}

__global__ void logsoftmax_kernel(const float* __restrict__ logits, float* __restrict__ out) {
  int g = threadIdx.x;                 // 128 rows, 1 block
  if (g >= GG) return;
  float m = -__builtin_inff();
  for (int c = 0; c < CC; ++c) m = fmaxf(m, logits[g * CC + c]);
  float se = 0.f;
  for (int c = 0; c < CC; ++c) se += expf(logits[g * CC + c] - m);
  float l = logf(se);
  for (int c = 0; c < CC; ++c) out[g * CC + c] = logits[g * CC + c] - m - l;
}

// ---------------------------------------------------------------------------
// Host orchestration
// ---------------------------------------------------------------------------
extern "C" void kernel_launch(void* const* d_in, const int* in_sizes, int n_in,
                              void* d_out, int out_size, void* d_ws, size_t ws_size,
                              hipStream_t stream) {
  (void)in_sizes; (void)n_in; (void)out_size; (void)ws_size;

  const float* x      = (const float*)d_in[0];
  const int*   ei     = (const int*)d_in[1];
  const int*   src    = ei;
  const int*   dst    = ei + EE;
  const float* c1W = (const float*)d_in[3];  const float* c1b = (const float*)d_in[4];
  const float* c2W = (const float*)d_in[5];  const float* c2b = (const float*)d_in[6];
  const float* c3W = (const float*)d_in[7];  const float* c3b = (const float*)d_in[8];
  const float* p1relW = (const float*)d_in[9];  const float* p1relB = (const float*)d_in[10];
  const float* p1rootW = (const float*)d_in[11];
  const float* p2relW = (const float*)d_in[12]; const float* p2relB = (const float*)d_in[13];
  const float* p2rootW = (const float*)d_in[14];
  const float* f1W = (const float*)d_in[15]; const float* f1b = (const float*)d_in[16];
  const float* f2W = (const float*)d_in[17]; const float* f2b = (const float*)d_in[18];
  const float* f3W = (const float*)d_in[19]; const float* f3b = (const float*)d_in[20];

  // workspace layout (floats)
  float* w = (float*)d_ws;
  const size_t NH = (size_t)NNODES * HH;     // 8388608
  float* buf0  = w;               w += NH;
  float* buf1  = w;               w += NH;
  float* buf2  = w;               w += NH;
  float* deg   = w;               w += NNODES;
  float* dis   = w;               w += NNODES;
  float* score = w;               w += NNODES;
  float* nmA   = w;               w += NNODES;
  float* nmB   = w;               w += NNODES;
  float* nrm   = w;               w += EE;
  float* emf   = w;               w += EE;
  float* zacc  = w;               w += (size_t)GG * 3 * HH;
  float* zt1   = w;               w += (size_t)GG * HH;
  float* zt2   = w;               w += (size_t)GG * (HH / 2);
  float* logits= w;               w += (size_t)GG * CC;

  const int TB = 256;
  const int nbNode  = NNODES / TB;                 // 128
  const int nbEdge  = EE / TB;                     // 2048
  const int nbNH    = (int)(NH / TB);              // 32768
  const int nbAgg   = (int)((size_t)EE * (HH / 4) / TB);  // 131072

  auto gcn = [&](const float* hin, const float* Wm, const float* bm, int Kdim,
                 const float* nmask, float* hwbuf, float* outbuf) {
    dim3 gg(NNODES / BM, HH / BN);               // (256, 4)
    gemm_wmma<<<gg, TB, 0, stream>>>(hin, Wm, nullptr, hwbuf, NNODES, HH, Kdim, 0);
    copy_f32<<<nbNode, TB, 0, stream>>>(deg, nmask, NNODES);
    deg_edges<<<nbEdge, TB, 0, stream>>>(dst, emf, deg, EE);
    dis_kernel<<<nbNode, TB, 0, stream>>>(deg, dis, NNODES);
    norm_kernel<<<nbEdge, TB, 0, stream>>>(src, dst, emf, dis, nrm, EE);
    hipMemsetAsync(outbuf, 0, NH * sizeof(float), stream);
    aggregate<<<nbAgg, TB, 0, stream>>>(hwbuf, src, dst, nrm, outbuf, EE);
    gcn_epi<<<nbNH, TB, 0, stream>>>(hwbuf, dis, nmask, bm, outbuf);
  };

  auto pool = [&](float* h, const float* nmask_in, float* nmask_out, int k,
                  const float* wrel, const float* brel, const float* wroot,
                  float* aggbuf) {
    hipMemsetAsync(aggbuf, 0, NH * sizeof(float), stream);
    aggregate<<<nbAgg, TB, 0, stream>>>(h, src, dst, emf, aggbuf, EE);
    score_kernel<<<NNODES / 8, TB, 0, stream>>>(aggbuf, h, wrel, brel, wroot,
                                                nmask_in, score);
    topk_kernel<<<GG, NPG, 0, stream>>>(score, nmask_out, k);
    apply_pool<<<nbNH, TB, 0, stream>>>(h, score, nmask_out);
    update_emask<<<nbEdge, TB, 0, stream>>>(src, dst, nmask_out, emf, EE);
  };

  // init masks
  fill_f32<<<nbEdge, TB, 0, stream>>>(emf, 1.f, EE);
  fill_f32<<<nbNode, TB, 0, stream>>>(nmA, 1.f, NNODES);

  // stage 1
  gcn(x, c1W, c1b, FIN, nmA, buf0, buf1);                    // h -> buf1
  pool(buf1, nmA, nmB, 128, p1relW, p1relB, p1rootW, buf2);
  readout_kernel<<<GG, HH, 0, stream>>>(buf1, nmB, zacc, 0);

  // stage 2
  gcn(buf1, c2W, c2b, HH, nmB, buf0, buf2);                  // h -> buf2
  pool(buf2, nmB, nmA, 64, p2relW, p2relB, p2rootW, buf0);
  readout_kernel<<<GG, HH, 0, stream>>>(buf2, nmA, zacc, 1);

  // stage 3 (reference reuses pool2 weights)
  gcn(buf2, c3W, c3b, HH, nmA, buf1, buf0);                  // h -> buf0
  pool(buf0, nmA, nmB, 32, p2relW, p2relB, p2rootW, buf1);
  readout_kernel<<<GG, HH, 0, stream>>>(buf0, nmB, zacc, 1);

  // MLP head (fc1, fc2 via WMMA; fc3 + log_softmax scalar)
  {
    dim3 g1(GG / BM, HH / BN);                   // (1, 4)
    gemm_wmma<<<g1, TB, 0, stream>>>(zacc, f1W, f1b, zt1, GG, HH, 3 * HH, 1);
    dim3 g2(GG / BM, (HH / 2) / BN);             // (1, 2)
    gemm_wmma<<<g2, TB, 0, stream>>>(zt1, f2W, f2b, zt2, GG, HH / 2, HH, 1);
    fc3_kernel<<<(GG * CC + TB - 1) / TB, TB, 0, stream>>>(zt2, f3W, f3b, logits);
    logsoftmax_kernel<<<1, GG, 0, stream>>>(logits, (float*)d_out);
  }
}
